// HetGraphLayer_79809082294311
// MI455X (gfx1250) — matrix operand order repaired and checked by reading, not compiled, not run on hardware
//
#include <hip/hip_runtime.h>
#include <hip/hip_bf16.h>
#include <math.h>

typedef __attribute__((ext_vector_type(16))) __bf16 v16bf;
typedef __attribute__((ext_vector_type(8)))  float  v8f;

#define D 128
#define NREL 3

// ---------------------------------------------------------------------------
// Build a 16x32 bf16 A-fragment from a row-major f32 row, fused scale.
// lane<16 : elements 0-7 = K kb..kb+7, elements 8-15 = K kb+16..kb+23 (kb = kt*32)
// lane>=16: same with kb += 8  (caller passes kb already adjusted)
// ---------------------------------------------------------------------------
__device__ __forceinline__ v16bf load_a_frag_f32(const float* __restrict__ row,
                                                 int kb, float scale) {
  float4 a0 = *(const float4*)(row + kb);
  float4 a1 = *(const float4*)(row + kb + 4);
  float4 b0 = *(const float4*)(row + kb + 16);
  float4 b1 = *(const float4*)(row + kb + 20);
  v16bf a;
  a[0] = (__bf16)(a0.x * scale); a[1] = (__bf16)(a0.y * scale);
  a[2] = (__bf16)(a0.z * scale); a[3] = (__bf16)(a0.w * scale);
  a[4] = (__bf16)(a1.x * scale); a[5] = (__bf16)(a1.y * scale);
  a[6] = (__bf16)(a1.z * scale); a[7] = (__bf16)(a1.w * scale);
  a[8]  = (__bf16)(b0.x * scale); a[9]  = (__bf16)(b0.y * scale);
  a[10] = (__bf16)(b0.z * scale); a[11] = (__bf16)(b0.w * scale);
  a[12] = (__bf16)(b1.x * scale); a[13] = (__bf16)(b1.y * scale);
  a[14] = (__bf16)(b1.z * scale); a[15] = (__bf16)(b1.w * scale);
  return a;
}

// ---------------------------------------------------------------------------
// Pack W[3][128][128] and Wa1[128][128] (f32, row-major K x N) into the
// per-lane WMMA-B bf16 layout: [mat][kt][nt][lane][16] contiguous.
//   col = nt*16 + (lane&15);  k = kt*32 + (lane>>4)*8 + (i>>3)*16 + (i&7)
// ---------------------------------------------------------------------------
__global__ void pack_weights(const float* __restrict__ W,
                             const float* __restrict__ Wa1,
                             __bf16* __restrict__ Wp,
                             __bf16* __restrict__ Wa1p) {
  int t = blockIdx.x * blockDim.x + threadIdx.x;      // 0 .. 4*16384-1
  int mat  = t >> 14;
  int idx  = t & 16383;
  int i    = idx & 15;
  int lane = (idx >> 4) & 31;
  int nt   = (idx >> 9) & 7;
  int kt   = (idx >> 12) & 3;
  int k    = kt * 32 + (lane >> 4) * 8 + (i >> 3) * 16 + (i & 7);
  int col  = nt * 16 + (lane & 15);
  if (mat < NREL) {
    Wp[(size_t)mat * 16384 + idx] = (__bf16)W[(size_t)mat * 16384 + k * D + col];
  } else {
    Wa1p[idx] = (__bf16)Wa1[k * D + col];
  }
}

// ---------------------------------------------------------------------------
// Degree histograms for all relations (u32 hardware atomics, L2 resident).
// ---------------------------------------------------------------------------
__global__ void degree_kernel(const int* __restrict__ src,
                              const int* __restrict__ dst,
                              unsigned* __restrict__ degO,
                              unsigned* __restrict__ degI,
                              int N, int E) {
  int t = blockIdx.x * blockDim.x + threadIdx.x;
  if (t >= NREL * E) return;
  int r = t / E;
  atomicAdd(&degO[(size_t)r * N + src[t]], 1u);
  atomicAdd(&degI[(size_t)r * N + dst[t]], 1u);
}

// ---------------------------------------------------------------------------
// h = bf16(x * rsqrt(deg_out)) @ W_r   (M=N nodes, K=128, N=128)
// One wave per 16-row tile, 8 accumulator tiles cover all 128 columns.
// ---------------------------------------------------------------------------
__global__ void __launch_bounds__(128)
gemm_xw(const float* __restrict__ x,
        const unsigned* __restrict__ degO_r,
        const __bf16* __restrict__ Wp_r,
        float* __restrict__ h, int N, int tiles) {
  int lane = threadIdx.x & 31;
  int tile = blockIdx.x * 4 + (threadIdx.x >> 5);
  if (tile >= tiles) return;                 // wave-uniform skip, EXEC stays full

  int row = tile * 16 + (lane & 15);
  if (row >= N) row = N - 1;
  unsigned dv = degO_r[row]; if (dv == 0u) dv = 1u;
  float scale = rsqrtf((float)dv);
  const float* xr = x + (size_t)row * D;

  v8f acc[8];
#pragma unroll
  for (int nt = 0; nt < 8; ++nt) acc[nt] = (v8f){0,0,0,0,0,0,0,0};

  int khalf = (lane >> 4) * 8;
#pragma unroll
  for (int kt = 0; kt < 4; ++kt) {
    v16bf a = load_a_frag_f32(xr, kt * 32 + khalf, scale);
    const __bf16* wpk = Wp_r + ((size_t)kt * 8) * 512 + (size_t)lane * 16;
#pragma unroll
    for (int nt = 0; nt < 8; ++nt) {
      v16bf bfrag = *(const v16bf*)(wpk + (size_t)nt * 512);
      acc[nt] = __builtin_amdgcn_wmma_f32_16x16x32_bf16(
          false, a, false, bfrag, (short)0, acc[nt], false, false);
    }
  }

  int rbase = tile * 16 + (lane >> 4) * 8;
  int colb  = lane & 15;
#pragma unroll
  for (int nt = 0; nt < 8; ++nt) {
#pragma unroll
    for (int j = 0; j < 8; ++j) {
      int rr = rbase + j;
      if (rr < N) h[(size_t)rr * D + nt * 16 + colb] = acc[nt][j];
    }
  }
}

// ---------------------------------------------------------------------------
// One wave per edge: gather h[src] row (float4/lane), scatter-add to rel_r[dst]
// with hardware f32 atomics (L2-resident destination array).
// ---------------------------------------------------------------------------
__global__ void __launch_bounds__(256)
scatter_edges(const float* __restrict__ h,
              const int* __restrict__ src_r,
              const int* __restrict__ dst_r,
              float* __restrict__ rel_r, int E) {
  int wid  = blockIdx.x * (blockDim.x >> 5) + (threadIdx.x >> 5);
  int lane = threadIdx.x & 31;
  if (wid >= E) return;
  int s = src_r[wid];
  int d = dst_r[wid];
  float4 v = *(const float4*)(h + (size_t)s * D + lane * 4);
  float* p = rel_r + (size_t)d * D + lane * 4;
  unsafeAtomicAdd(p + 0, v.x);
  unsafeAtomicAdd(p + 1, v.y);
  unsafeAtomicAdd(p + 2, v.z);
  unsafeAtomicAdd(p + 3, v.w);
}

// ---------------------------------------------------------------------------
// rel = agg * rsqrt(deg_in) + b   (in place, float4 per thread)
// ---------------------------------------------------------------------------
__global__ void finalize_rel(float* __restrict__ rel,
                             const unsigned* __restrict__ degI,
                             const float* __restrict__ b, int N) {
  long long t = (long long)blockIdx.x * blockDim.x + threadIdx.x;
  if (t >= (long long)NREL * N * 32) return;
  int c4 = (int)(t & 31);
  long long nr = t >> 5;
  int n = (int)(nr % N);
  int r = (int)(nr / N);
  unsigned dv = degI[(size_t)r * N + n]; if (dv == 0u) dv = 1u;
  float s = rsqrtf((float)dv);
  size_t off = ((size_t)r * N + n) * D + c4 * 4;
  float4 v  = *(float4*)(rel + off);
  float4 bb = *(const float4*)(b + r * D + c4 * 4);
  v.x = v.x * s + bb.x;  v.y = v.y * s + bb.y;
  v.z = v.z * s + bb.z;  v.w = v.w * s + bb.w;
  *(float4*)(rel + off) = v;
}

// ---------------------------------------------------------------------------
// score[r][n] = tanh(rel[r][n] @ Wa1 + ba1) . wa2  via WMMA + cross-lane reduce
// One wave per (relation, 16-row tile).
// ---------------------------------------------------------------------------
__global__ void __launch_bounds__(128)
attn_score(const float* __restrict__ rel,
           const __bf16* __restrict__ Wa1p,
           const float* __restrict__ ba1,
           const float* __restrict__ wa2,
           float* __restrict__ score, int N, int tiles) {
  int lane = threadIdx.x & 31;
  int w = blockIdx.x * 4 + (threadIdx.x >> 5);
  if (w >= NREL * tiles) return;
  int r = w / tiles;
  int tile = w - r * tiles;

  int row = tile * 16 + (lane & 15);
  if (row >= N) row = N - 1;
  const float* xr = rel + ((size_t)r * N + row) * D;

  float ba[8], wa[8];
#pragma unroll
  for (int nt = 0; nt < 8; ++nt) {
    int col = nt * 16 + (lane & 15);
    ba[nt] = ba1[col];
    wa[nt] = wa2[col];
  }

  v8f acc[8];
#pragma unroll
  for (int nt = 0; nt < 8; ++nt) acc[nt] = (v8f){0,0,0,0,0,0,0,0};

  int khalf = (lane >> 4) * 8;
#pragma unroll
  for (int kt = 0; kt < 4; ++kt) {
    v16bf a = load_a_frag_f32(xr, kt * 32 + khalf, 1.0f);
    const __bf16* wpk = Wa1p + ((size_t)kt * 8) * 512 + (size_t)lane * 16;
#pragma unroll
    for (int nt = 0; nt < 8; ++nt) {
      v16bf bfrag = *(const v16bf*)(wpk + (size_t)nt * 512);
      acc[nt] = __builtin_amdgcn_wmma_f32_16x16x32_bf16(
          false, a, false, bfrag, (short)0, acc[nt], false, false);
    }
  }

  // per-row: sum over 128 cols of tanh(t + ba1)*wa2; cols are spread over
  // the 16 lanes of each half-wave x 8 acc tiles -> butterfly reduce.
#pragma unroll
  for (int j = 0; j < 8; ++j) {
    float p = 0.f;
#pragma unroll
    for (int nt = 0; nt < 8; ++nt)
      p += tanhf(acc[nt][j] + ba[nt]) * wa[nt];
    p += __shfl_xor(p, 1, 32);
    p += __shfl_xor(p, 2, 32);
    p += __shfl_xor(p, 4, 32);
    p += __shfl_xor(p, 8, 32);
    if ((lane & 15) == 0) {
      int rr = tile * 16 + (lane >> 4) * 8 + j;
      if (rr < N) score[(size_t)r * N + rr] = p;
    }
  }
}

// ---------------------------------------------------------------------------
// out[n] = sum_r softmax_r(score[.,n]) * rel[r][n]   (float4 per thread)
// ---------------------------------------------------------------------------
__global__ void attn_output(const float* __restrict__ rel,
                            const float* __restrict__ score,
                            float* __restrict__ out, int N) {
  int t = blockIdx.x * blockDim.x + threadIdx.x;
  if (t >= N * 32) return;
  int n = t >> 5;
  int c4 = t & 31;
  float s0 = score[n], s1 = score[(size_t)N + n], s2 = score[(size_t)2 * N + n];
  float m = fmaxf(s0, fmaxf(s1, s2));
  float e0 = __expf(s0 - m), e1 = __expf(s1 - m), e2 = __expf(s2 - m);
  float inv = 1.0f / (e0 + e1 + e2);
  e0 *= inv; e1 *= inv; e2 *= inv;
  size_t off = (size_t)n * D + c4 * 4;
  float4 v0 = *(const float4*)(rel + off);
  float4 v1 = *(const float4*)(rel + (size_t)N * D + off);
  float4 v2 = *(const float4*)(rel + (size_t)2 * N * D + off);
  float4 o;
  o.x = e0 * v0.x + e1 * v1.x + e2 * v2.x;
  o.y = e0 * v0.y + e1 * v1.y + e2 * v2.y;
  o.z = e0 * v0.z + e1 * v1.z + e2 * v2.z;
  o.w = e0 * v0.w + e1 * v1.w + e2 * v2.w;
  *(float4*)(out + off) = o;
}

// ---------------------------------------------------------------------------
extern "C" void kernel_launch(void* const* d_in, const int* in_sizes, int n_in,
                              void* d_out, int out_size, void* d_ws, size_t ws_size,
                              hipStream_t stream) {
  const float* x   = (const float*)d_in[0];
  const int*   src = (const int*)  d_in[1];
  const int*   dst = (const int*)  d_in[2];
  const float* W   = (const float*)d_in[3];
  const float* b   = (const float*)d_in[4];
  const float* Wa1 = (const float*)d_in[5];
  const float* ba1 = (const float*)d_in[6];
  const float* wa2 = (const float*)d_in[7];

  const int N = in_sizes[0] / D;        // 100000
  const int E = in_sizes[1] / NREL;     // 640000

  auto al = [](size_t v) { return (v + 255) & ~(size_t)255; };
  char* ws = (char*)d_ws;
  size_t off = 0;
  float*    rel   = (float*)(ws + off);    off += al((size_t)NREL * N * D * 4);
  float*    h     = (float*)(ws + off);    off += al((size_t)N * D * 4);
  unsigned* degO  = (unsigned*)(ws + off); off += al((size_t)NREL * N * 4);
  unsigned* degI  = (unsigned*)(ws + off); off += al((size_t)NREL * N * 4);
  __bf16*   Wp    = (__bf16*)(ws + off);   off += al((size_t)NREL * D * D * 2);
  __bf16*   Wa1p  = (__bf16*)(ws + off);   off += al((size_t)D * D * 2);
  float*    score = (float*)(ws + off);    off += al((size_t)NREL * N * 4);
  (void)ws_size; (void)n_in; (void)out_size;

  hipMemsetAsync(rel,  0, (size_t)NREL * N * D * 4, stream);
  hipMemsetAsync(degO, 0, (size_t)NREL * N * 4, stream);
  hipMemsetAsync(degI, 0, (size_t)NREL * N * 4, stream);

  pack_weights<<<(4 * D * D + 255) / 256, 256, 0, stream>>>(W, Wa1, Wp, Wa1p);
  degree_kernel<<<(NREL * E + 255) / 256, 256, 0, stream>>>(src, dst, degO, degI, N, E);

  const int tiles = (N + 15) / 16;
  const int gemm_blocks = (tiles + 3) / 4;
  const int scat_blocks = (E + 7) / 8;
  for (int r = 0; r < NREL; ++r) {
    gemm_xw<<<gemm_blocks, 128, 0, stream>>>(
        x, degO + (size_t)r * N, Wp + (size_t)r * D * D, h, N, tiles);
    scatter_edges<<<scat_blocks, 256, 0, stream>>>(
        h, src + (size_t)r * E, dst + (size_t)r * E, rel + (size_t)r * N * D, E);
  }

  finalize_rel<<<(int)(((long long)NREL * N * 32 + 255) / 256), 256, 0, stream>>>(
      rel, degI, b, N);
  attn_score<<<(NREL * tiles + 3) / 4, 128, 0, stream>>>(
      rel, Wa1p, ba1, wa2, score, N, tiles);
  attn_output<<<(N * 32 + 255) / 256, 256, 0, stream>>>(
      rel, score, (float*)d_out, N);
}
